// NonLocal_6227702579963
// MI455X (gfx1250) — compile-verified, matrix operand order checked by reading, not run on hardware
//
#include <hip/hip_runtime.h>
#include <hip/hip_bf16.h>
#include <stdint.h>

// ---------------------------------------------------------------------------
// CDNA5 WMMA types (wave32): D(16x16 f32, v8f) = A(16x32 bf16) x B + C
// ---------------------------------------------------------------------------
typedef __attribute__((ext_vector_type(16))) __bf16 v16bf;
typedef __attribute__((ext_vector_type(8)))  float  v8f;
typedef __attribute__((ext_vector_type(4)))  int    v4i;

typedef __attribute__((address_space(1))) v4i* gptr_v4i;   // global v4i*
typedef __attribute__((address_space(3))) v4i* lptr_v4i;   // LDS v4i*

#define TBM 128
#define TBN 128
#define TBK 32

#ifndef __has_builtin
#define __has_builtin(x) 0
#endif

#if __has_builtin(__builtin_amdgcn_global_load_async_to_lds_b128)
#define HAS_ASYNC_LDS 1
// (global src, lds dst, imm offset, imm cpol); IOFFSET is added to both addrs
#define ASYNC_CP16(g, l, off)                                                  \
  __builtin_amdgcn_global_load_async_to_lds_b128(                              \
      (gptr_v4i)(const void*)(g), (lptr_v4i)(void*)(l), (off), 0)
#else
#define HAS_ASYNC_LDS 0
#define ASYNC_CP16(g, l, off)                                                  \
  (*(uint4*)((char*)(l) + (off)) =                                             \
       *(const uint4*)((const char*)(g) + (off)))
#endif

__device__ __forceinline__ void wait_async0() {
#if HAS_ASYNC_LDS
#if __has_builtin(__builtin_amdgcn_s_wait_asynccnt)
  __builtin_amdgcn_s_wait_asynccnt(0);
#else
  asm volatile("s_wait_asynccnt 0x0" ::: "memory");
#endif
#endif
}

__device__ __forceinline__ unsigned short f32_to_bf16(float f) {
  union { float f; unsigned int u; } c; c.f = f;
  unsigned int u = c.u;
  unsigned int r = u + 0x7FFFu + ((u >> 16) & 1u);   // round-to-nearest-even
  return (unsigned short)(r >> 16);
}
__device__ __forceinline__ unsigned short to_bf16u(float f)         { return f32_to_bf16(f); }
__device__ __forceinline__ unsigned short to_bf16u(unsigned short s){ return s; }

union Frag16 { v16bf v; uint4 q[2]; };

// ---------------------------------------------------------------------------
// bf16 WMMA GEMM: C[M,N] = A[M,K] * B^T[N,K].
// 128x128 block tile, 128 threads = 4 waves (2x2), 64x64 per wave
// (4x4 WMMA frags -> 8 fragment loads per 16 WMMAs), BK=32,
// double-buffered LDS filled by async global->LDS b128 copies.
// Epilogues:
//   EPI=0: f32 store C0[m*ldc+n]                          (attn^T)
//   EPI=1: bf16 store to y_t layout C0[p*256+ic]          (y ready for mask GEMM)
//   EPI=2: conv scatter: phi_t(C0)/theta_t(C1)/g_v(C2) via the view remap
//   EPI=3: f32 store C0[m*ldc+n] = acc + Xadd[m*ldc+n]    (mask + residual)
// ---------------------------------------------------------------------------
template<int EPI>
__global__ __launch_bounds__(128)
void gemm_wmma(const unsigned short* __restrict__ A, int lda,
               const unsigned short* __restrict__ B, int ldb, int K,
               void* __restrict__ C0, void* __restrict__ C1, void* __restrict__ C2,
               const float* __restrict__ Xadd, int ldc)
{
  __shared__ alignas(16) unsigned short As[2][TBM][TBK];   // 2 x 8 KB
  __shared__ alignas(16) unsigned short Bs[2][TBN][TBK];   // 2 x 8 KB

  const int tid   = threadIdx.x;       // 0..127
  const int lane  = tid & 31;
  const int wave  = tid >> 5;          // 0..3
  const int waveM = wave >> 1;         // 0..1  -> 64 rows each
  const int waveN = wave & 1;          // 0..1  -> 64 cols each
  const int blockM = blockIdx.y * TBM;
  const int blockN = blockIdx.x * TBN;

  const int lrow  = lane & 15;
  const int khalf = (lane >> 4) * 8;   // K-half striping per ISA A/B layout

  v8f acc[4][4];
  #pragma unroll
  for (int mi = 0; mi < 4; ++mi)
    #pragma unroll
    for (int ni = 0; ni < 4; ++ni)
      acc[mi][ni] = (v8f){0.f,0.f,0.f,0.f,0.f,0.f,0.f,0.f};

  const int nstages = K / TBK;

  // stage 0 prefetch: each thread fills one full 32-elem (64 B) tile row
  {
    size_t ga = (size_t)(blockM + tid) * (size_t)lda;
    size_t gb = (size_t)(blockN + tid) * (size_t)ldb;
    ASYNC_CP16(A + ga, &As[0][tid][0], 0);
    ASYNC_CP16(A + ga, &As[0][tid][0], 16);
    ASYNC_CP16(A + ga, &As[0][tid][0], 32);
    ASYNC_CP16(A + ga, &As[0][tid][0], 48);
    ASYNC_CP16(B + gb, &Bs[0][tid][0], 0);
    ASYNC_CP16(B + gb, &Bs[0][tid][0], 16);
    ASYNC_CP16(B + gb, &Bs[0][tid][0], 32);
    ASYNC_CP16(B + gb, &Bs[0][tid][0], 48);
  }

  for (int s = 0; s < nstages; ++s) {
    const int cur = s & 1;
    wait_async0();        // my async copies for stage s have landed in LDS
    __syncthreads();      // everyone's have; also: all reads of buf cur^1 done

    if (s + 1 < nstages) {
      const int k0 = (s + 1) * TBK;
      const int nxt = cur ^ 1;
      size_t ga = (size_t)(blockM + tid) * (size_t)lda + (size_t)k0;
      size_t gb = (size_t)(blockN + tid) * (size_t)ldb + (size_t)k0;
      ASYNC_CP16(A + ga, &As[nxt][tid][0], 0);
      ASYNC_CP16(A + ga, &As[nxt][tid][0], 16);
      ASYNC_CP16(A + ga, &As[nxt][tid][0], 32);
      ASYNC_CP16(A + ga, &As[nxt][tid][0], 48);
      ASYNC_CP16(B + gb, &Bs[nxt][tid][0], 0);
      ASYNC_CP16(B + gb, &Bs[nxt][tid][0], 16);
      ASYNC_CP16(B + gb, &Bs[nxt][tid][0], 32);
      ASYNC_CP16(B + gb, &Bs[nxt][tid][0], 48);
    }

    // fragments from LDS (2 x ds_load_b128 each): 8 frag loads / 16 WMMAs
    Frag16 af[4], bfr[4];
    #pragma unroll
    for (int mi = 0; mi < 4; ++mi) {
      int r = waveM * 64 + mi * 16 + lrow;
      af[mi].q[0] = *(const uint4*)&As[cur][r][khalf];
      af[mi].q[1] = *(const uint4*)&As[cur][r][khalf + 16];
    }
    #pragma unroll
    for (int ni = 0; ni < 4; ++ni) {
      int c = waveN * 64 + ni * 16 + lrow;
      bfr[ni].q[0] = *(const uint4*)&Bs[cur][c][khalf];
      bfr[ni].q[1] = *(const uint4*)&Bs[cur][c][khalf + 16];
    }
    #pragma unroll
    for (int mi = 0; mi < 4; ++mi)
      #pragma unroll
      for (int ni = 0; ni < 4; ++ni)
        acc[mi][ni] = __builtin_amdgcn_wmma_f32_16x16x32_bf16(
            false, af[mi].v, false, bfr[ni].v, (short)0, acc[mi][ni], false, false);
  }

  // epilogue: C/D layout -> element (m = v + 8*(lane>>4), n = lane&15)
  const int mrow0 = blockM + waveM * 64;
  const int ncol0 = blockN + waveN * 64;
  #pragma unroll
  for (int mi = 0; mi < 4; ++mi) {
    #pragma unroll
    for (int ni = 0; ni < 4; ++ni) {
      #pragma unroll
      for (int v = 0; v < 8; ++v) {
        int m = mrow0 + mi * 16 + v + 8 * (lane >> 4);
        int n = ncol0 + ni * 16 + (lane & 15);
        float val = acc[mi][ni][v];
        if constexpr (EPI == 0) {
          ((float*)C0)[(size_t)m * ldc + n] = val;
        } else if constexpr (EPI == 1) {
          // m = c in [0,512), n = i in [0,2048): y_t[p][ic], p=(c&1)*2048+i, ic=c>>1
          int p = (m & 1) * 2048 + n;
          ((unsigned short*)C0)[(size_t)p * 256 + (m >> 1)] = f32_to_bf16(val);
        } else if constexpr (EPI == 2) {
          // m in [0,768): 0-255 phi, 256-511 theta, 512-767 g ; n = p in [0,4096)
          int wsel = m >> 8, o = m & 255;
          int c  = 2 * o + (n >> 11);      // view channel in [0,512)
          int nn = n & 2047;               // view position in [0,2048)
          unsigned short bv = f32_to_bf16(val);
          if      (wsel == 0) ((unsigned short*)C0)[(size_t)nn * 512 + c]  = bv; // phi_t[n,c]
          else if (wsel == 1) ((unsigned short*)C1)[(size_t)nn * 512 + c]  = bv; // theta_t[n,c]
          else                ((unsigned short*)C2)[(size_t)c * 2048 + nn] = bv; // g_v[c,n]
        } else { // EPI == 3
          size_t idx = (size_t)m * ldc + n;
          ((float*)C0)[idx] = val + Xadd[idx];
        }
      }
    }
  }
}

// ---------------------------------------------------------------------------
// LDS-tiled transpose, output bf16: out[C][R] = cvt(in[R][C])
// block 256 = 32x8, 32x32 tile
// ---------------------------------------------------------------------------
template<typename TIN>
__global__ __launch_bounds__(256)
void transpose_to_bf16(const TIN* __restrict__ in, unsigned short* __restrict__ out,
                       int R, int C)
{
  __shared__ TIN t[32][33];
  const int bx = blockIdx.x * 32;             // along C
  const int by = blockIdx.y * 32;             // along R
  const int tx = threadIdx.x & 31;
  const int ty = threadIdx.x >> 5;            // 0..7
  #pragma unroll
  for (int i = 0; i < 4; ++i)
    t[ty + i * 8][tx] = in[(size_t)(by + ty + i * 8) * C + (bx + tx)];
  __syncthreads();
  #pragma unroll
  for (int i = 0; i < 4; ++i)
    out[(size_t)(bx + ty + i * 8) * R + (by + tx)] = to_bf16u(t[tx][ty + i * 8]);
}

// ---------------------------------------------------------------------------
// Column softmax of attn via its transposed storage; one block per row of
// attn^T (2048 f32), bf16 probabilities out.
// ---------------------------------------------------------------------------
__global__ __launch_bounds__(256)
void softmax_rows(const float* __restrict__ attnT, unsigned short* __restrict__ P)
{
  __shared__ float sm[256];
  const int j = blockIdx.x;
  const float* row = attnT + (size_t)j * 2048;
  float vals[8];
  float mx = -3.402823466e38f;
  #pragma unroll
  for (int i = 0; i < 8; ++i) {
    vals[i] = row[threadIdx.x + i * 256];
    mx = fmaxf(mx, vals[i]);
  }
  sm[threadIdx.x] = mx; __syncthreads();
  for (int s = 128; s > 0; s >>= 1) {
    if (threadIdx.x < s) sm[threadIdx.x] = fmaxf(sm[threadIdx.x], sm[threadIdx.x + s]);
    __syncthreads();
  }
  mx = sm[0]; __syncthreads();

  float sum = 0.f;
  #pragma unroll
  for (int i = 0; i < 8; ++i) { vals[i] = __expf(vals[i] - mx); sum += vals[i]; }
  sm[threadIdx.x] = sum; __syncthreads();
  for (int s = 128; s > 0; s >>= 1) {
    if (threadIdx.x < s) sm[threadIdx.x] += sm[threadIdx.x + s];
    __syncthreads();
  }
  float inv = 1.0f / sm[0];
  #pragma unroll
  for (int i = 0; i < 8; ++i)
    P[(size_t)j * 2048 + threadIdx.x + i * 256] = f32_to_bf16(vals[i] * inv);
}

__global__ __launch_bounds__(256)
void f32_to_bf16_kernel(const float* __restrict__ in, unsigned short* __restrict__ out, int n)
{
  int i = blockIdx.x * 256 + threadIdx.x;
  if (i < n) out[i] = f32_to_bf16(in[i]);
}

// ---------------------------------------------------------------------------
// Host launcher
// ---------------------------------------------------------------------------
extern "C" void kernel_launch(void* const* d_in, const int* in_sizes, int n_in,
                              void* d_out, int out_size, void* d_ws, size_t ws_size,
                              hipStream_t stream)
{
  (void)in_sizes; (void)n_in; (void)out_size; (void)ws_size;
  const float* x  = (const float*)d_in[0];  // (8,512,64,64)
  const float* wp = (const float*)d_in[1];  // (256,512)
  const float* wt = (const float*)d_in[2];  // (256,512)
  const float* wg = (const float*)d_in[3];  // (256,512)
  const float* wm = (const float*)d_in[4];  // (512,256)
  float* out = (float*)d_out;

  // workspace carve (~45 MB); per-batch buffers are reused across the loop
  char* ws = (char*)d_ws;
  unsigned short* w3   = (unsigned short*)ws; ws += (size_t)768 * 512 * 2;   // fused conv weights
  unsigned short* wmk  = (unsigned short*)ws; ws += (size_t)512 * 256 * 2;
  unsigned short* xt   = (unsigned short*)ws; ws += (size_t)4096 * 512 * 2;  // x^T bf16
  unsigned short* phiT = (unsigned short*)ws; ws += (size_t)2048 * 512 * 2;
  unsigned short* theT = (unsigned short*)ws; ws += (size_t)2048 * 512 * 2;
  unsigned short* gv   = (unsigned short*)ws; ws += (size_t)512 * 2048 * 2;
  unsigned short* yt   = (unsigned short*)ws; ws += (size_t)4096 * 256 * 2;  // y in [p][ic]
  unsigned short* P    = (unsigned short*)ws; ws += (size_t)2048 * 2048 * 2;
  unsigned short* Pt   = (unsigned short*)ws; ws += (size_t)2048 * 2048 * 2;
  float*          attn = (float*)ws;          ws += (size_t)2048 * 2048 * 4;

  const int nw = 256 * 512;
  f32_to_bf16_kernel<<<(nw + 255) / 256, 256, 0, stream>>>(wp, w3,          nw);
  f32_to_bf16_kernel<<<(nw + 255) / 256, 256, 0, stream>>>(wt, w3 + nw,     nw);
  f32_to_bf16_kernel<<<(nw + 255) / 256, 256, 0, stream>>>(wg, w3 + 2 * nw, nw);
  f32_to_bf16_kernel<<<(nw + 255) / 256, 256, 0, stream>>>(wm, wmk, 512 * 256);

  for (int b = 0; b < 8; ++b) {
    const float* xb = x   + (size_t)b * 512 * 4096;
    float*       ob = out + (size_t)b * 512 * 4096;

    // x^T (bf16): xt[p][c] = x[c][p]
    transpose_to_bf16<float><<<dim3(4096 / 32, 512 / 32), 256, 0, stream>>>(
        xb, xt, 512, 4096);

    // fused convs: M=768, N=4096, K=512 ; B^T = xt
    gemm_wmma<2><<<dim3(4096 / TBN, 768 / TBM), 128, 0, stream>>>(
        w3, 512, xt, 512, 512, phiT, theT, gv, nullptr, 0);

    // attn^T: M=N=2048, K=512 ; A=phi_t, B^T=theta_t (both [n,c])
    gemm_wmma<0><<<dim3(2048 / TBN, 2048 / TBM), 128, 0, stream>>>(
        phiT, 512, theT, 512, 512, attn, nullptr, nullptr, nullptr, 2048);

    // softmax over original axis=1 == rows of attn^T
    softmax_rows<<<2048, 256, 0, stream>>>(attn, P);

    // Pt[i][j] = P[j][i]
    transpose_to_bf16<unsigned short><<<dim3(2048 / 32, 2048 / 32), 256, 0, stream>>>(
        P, Pt, 2048, 2048);

    // y: M=512(c), N=2048(i), K=2048(j) ; B^T = Pt ; epilogue stores y_t[p][ic]
    gemm_wmma<1><<<dim3(2048 / TBN, 512 / TBM), 128, 0, stream>>>(
        gv, 2048, Pt, 2048, 2048, yt, nullptr, nullptr, nullptr, 0);

    // mask + residual: M=512(oc), N=4096(p), K=256(ic) ; B^T = y_t
    gemm_wmma<3><<<dim3(4096 / TBN, 512 / TBM), 128, 0, stream>>>(
        wmk, 256, yt, 256, 256, ob, nullptr, nullptr, xb, 4096);
  }
}
